// LSHAttention_8366596293015
// MI455X (gfx1250) — matrix operand order repaired and checked by reading, compiled, not run on hardware
//
#include <hip/hip_runtime.h>
#include <hip/hip_bf16.h>
#include <math.h>

// ---- problem constants (from reference) ----
#define NB_   16      // batch
#define NS_   4096    // sequence
#define ND_   64      // head dim
#define NH_   8       // hash rounds
#define NBKT_ 64      // buckets per hash (= S / BUCKET_SIZE)
#define NCH_  512     // chunks per batch (= NH_ * NBKT_)
#define BSZ_  64      // bucket/chunk size
#define SELF_ATTN_VAL (-5.0e4f)

typedef __attribute__((ext_vector_type(16))) _Float16 v16h;
typedef __attribute__((ext_vector_type(8)))  _Float16 v8h;
typedef __attribute__((ext_vector_type(8)))  float    v8f;

// ------------------------------------------------------------------
// Fragment loaders matching CDNA5 16-bit WMMA VGPR layouts (ISA 7.12.2)
// A 16x32 (MxK): lane<16 -> K 0..7 / 16..23 ; lane>=16 -> K 8..15 / 24..31
// element (row m, K k) lives at base + m*stride + k  (row-major f16)
__device__ __forceinline__ v16h ldsA(const _Float16* base, int stride, int lane, int koff) {
    int m = lane & 15, half = lane >> 4;
    const _Float16* p = base + m * stride + koff;
    v8h lo = *(const v8h*)(p + half * 8);
    v8h hi = *(const v8h*)(p + 16 + half * 8);
    v16h r;
#pragma unroll
    for (int i = 0; i < 8; ++i) { r[i] = lo[i]; r[i + 8] = hi[i]; }
    return r;
}
// B 32x16 (KxN): lane<16 -> N=lane, K 0..7 / 16..23 ; lane>=16 -> N=lane-16, K 8..15 / 24..31
// element (K k, N n) lives at base + n*stride + k  (N-major rows of contiguous K)
__device__ __forceinline__ v16h ldsB(const _Float16* base, int stride, int lane, int koff) {
    int n = lane & 15, half = lane >> 4;
    const _Float16* p = base + n * stride + koff;
    v8h lo = *(const v8h*)(p + half * 8);
    v8h hi = *(const v8h*)(p + 16 + half * 8);
    v16h r;
#pragma unroll
    for (int i = 0; i < 8; ++i) { r[i] = lo[i]; r[i + 8] = hi[i]; }
    return r;
}

// ------------------------------------------------------------------
// Kernel 1: LSH hashing.  bucket[b,h,t] = argmax_i [r_i, -r_i], r = qk . rot
// grid (S/256, B), block 256.  Rotations (64 x 8 x 32 f32 = 64KB) cached in LDS.
__global__ __launch_bounds__(256) void lsh_hash_kernel(
    const float* __restrict__ qk, const float* __restrict__ rot,
    int* __restrict__ bkt) {
    __shared__ float sRot[ND_ * NH_ * 32];   // 64KB: [f][h][i] -> f*256 + h*32 + i
    const int tid = threadIdx.x;
    const int b = blockIdx.y;
    const int t = blockIdx.x * 256 + tid;
#pragma unroll
    for (int i = 0; i < (ND_ * NH_ * 32) / 256; ++i)
        sRot[i * 256 + tid] = rot[i * 256 + tid];
    __syncthreads();

    float q[ND_];
    const float* qrow = qk + ((size_t)b * NS_ + t) * ND_;
#pragma unroll
    for (int f = 0; f < ND_; ++f) q[f] = qrow[f];

    for (int h = 0; h < NH_; ++h) {
        float mx = -3.0e38f, mn = 3.0e38f;
        int ai = 0, mi = 0;
        for (int i = 0; i < 32; ++i) {
            float s = 0.0f;
            const float* rp = &sRot[h * 32 + i];
#pragma unroll
            for (int f = 0; f < ND_; ++f) s = fmaf(q[f], rp[f * 256], s);
            if (s > mx) { mx = s; ai = i; }
            if (s < mn) { mn = s; mi = i; }
        }
        int bucket = (mx >= -mn) ? ai : (32 + mi);   // argmax over [r, -r]
        bkt[((size_t)b * NH_ + h) * NS_ + t] = bucket;
    }
}

// ------------------------------------------------------------------
// Kernel 2: stable counting sort per (b,h) slice by (bucket, position).
// grid (H, B), block 64 (one thread owns one bucket).
__global__ __launch_bounds__(64) void lsh_sort_kernel(
    const int* __restrict__ bkt, int* __restrict__ st) {
    __shared__ int sB[NS_];        // 16KB
    __shared__ int sOff[NBKT_ + 1];
    const int tid = threadIdx.x;               // bucket owned by this thread
    const int h = blockIdx.x, b = blockIdx.y;
    const int* src = bkt + ((size_t)b * NH_ + h) * NS_;
    for (int i = tid; i < NS_; i += 64) sB[i] = src[i];
    __syncthreads();
    int cnt = 0;
    for (int t = 0; t < NS_; ++t) cnt += (sB[t] == tid);
    sOff[tid + 1] = cnt;
    __syncthreads();
    if (tid == 0) {
        sOff[0] = 0;
        for (int i = 1; i <= NBKT_; ++i) sOff[i] += sOff[i - 1];
    }
    __syncthreads();
    int off = sOff[tid];
    int* dst = st + ((size_t)b * NH_ + h) * NS_;
    for (int t = 0; t < NS_; ++t)
        if (sB[t] == tid) dst[off++] = t;      // stable: position order
}

// ------------------------------------------------------------------
// Kernel 3: chunked attention.  One workgroup (4 waves / 128 threads) per chunk.
// grid (512 chunks, B).  WMMA f32_16x16x32_f16 for QK^T and P.V.
__global__ __launch_bounds__(128) void lsh_attn_kernel(
    const float* __restrict__ qk, const float* __restrict__ v,
    const int* __restrict__ st, float* __restrict__ oh,
    float* __restrict__ lseh) {
    __shared__ _Float16 sQ[BSZ_ * ND_];          // 8KB  q rows (sorted order)
    __shared__ _Float16 sKP[2 * BSZ_ * ND_];     // 16KB K (j-major) then reused as P (row-major)
    __shared__ _Float16 sVT[ND_ * 2 * BSZ_];     // 16KB V transposed: [d][j]
    __shared__ float    sDO[BSZ_ * 2 * BSZ_];    // 32KB dots, first 64x64 reused as O
    __shared__ int      sTq[BSZ_];
    __shared__ int      sTk[2 * BSZ_];
    __shared__ float    sLse[BSZ_];

    const int tid = threadIdx.x;
    const int lane = tid & 31;
    const int wv = tid >> 5;                     // wave id 0..3 == M-tile
    const int c = blockIdx.x;                    // chunk 0..511
    const int b = blockIdx.y;
    const int cp = (c + NCH_ - 1) % NCH_;        // look-one-back (wraps whole axis)
    const int h = c >> 6;                        // 64 chunks per hash round
    const int* stb = st + (size_t)b * (NH_ * NS_);

    // ---- load q (64 rows), k normalized + v transposed (128 rows) ----
    if (tid < BSZ_) {
        int tq = stb[c * BSZ_ + tid];
        sTq[tid] = tq;
        const float* src = qk + ((size_t)b * NS_ + tq) * ND_;
#pragma unroll
        for (int e = 0; e < ND_; ++e) sQ[tid * ND_ + e] = (_Float16)src[e];
    }
    {
        int j = tid;                             // 0..127 key slot
        int tk = stb[((j < BSZ_) ? c : cp) * BSZ_ + (j & (BSZ_ - 1))];
        sTk[j] = tk;
        const float* ksrc = qk + ((size_t)b * NS_ + tk) * ND_;
        const float* vsrc = v + ((size_t)b * NS_ + tk) * ND_;
        float ss = 0.0f;
#pragma unroll
        for (int e = 0; e < ND_; ++e) { float x = ksrc[e]; ss = fmaf(x, x, ss); }
        float inv = 1.0f / fmaxf(sqrtf(ss), 1e-12f);
#pragma unroll
        for (int e = 0; e < ND_; ++e) {
            sKP[j * ND_ + e] = (_Float16)(ksrc[e] * inv);   // normalized K
            sVT[e * (2 * BSZ_) + j] = (_Float16)vsrc[e];    // V^T
        }
    }
    __syncthreads();

    // ---- dots = Q K^T : per wave 16 rows x 128 cols, K-dim 64 = 2x32 ----
    {
        v8f acc[8];
#pragma unroll
        for (int nt = 0; nt < 8; ++nt) acc[nt] = (v8f)(0.0f);
#pragma unroll
        for (int ks = 0; ks < 2; ++ks) {
            v16h a = ldsA(sQ + wv * 16 * ND_, ND_, lane, ks * 32);
#pragma unroll
            for (int nt = 0; nt < 8; ++nt) {
                v16h bb = ldsB(sKP + nt * 16 * ND_, ND_, lane, ks * 32);
                acc[nt] = __builtin_amdgcn_wmma_f32_16x16x32_f16(
                    false, a, false, bb, (short)0, acc[nt], false, false);
            }
        }
        int row0 = wv * 16 + (lane >> 4) * 8;
        int col0 = lane & 15;
#pragma unroll
        for (int nt = 0; nt < 8; ++nt)
#pragma unroll
            for (int r = 0; r < 8; ++r)
                sDO[(row0 + r) * (2 * BSZ_) + nt * 16 + col0] = acc[nt][r];
    }
    __syncthreads();

    // ---- mask + softmax (f32), write P (f16, reuses K buffer) ----
    if (tid < BSZ_) {
        const float scale = 0.125f;              // D^-0.5
        int tq = sTq[tid];
        float mx = -3.0e38f;
        for (int j = 0; j < 2 * BSZ_; ++j) {
            float x = sDO[tid * (2 * BSZ_) + j] * scale;
            if (tq == sTk[j]) x = SELF_ATTN_VAL;
            mx = fmaxf(mx, x);
        }
        float sum = 0.0f;
        for (int j = 0; j < 2 * BSZ_; ++j) {
            float x = sDO[tid * (2 * BSZ_) + j] * scale;
            if (tq == sTk[j]) x = SELF_ATTN_VAL;
            float p = expf(x - mx);
            sum += p;
            sKP[tid * (2 * BSZ_) + j] = (_Float16)p;   // unnormalized; /sum after GEMM
        }
        sLse[tid] = mx + logf(sum);
    }
    __syncthreads();

    // ---- O = P V : per wave 16 rows x 64 cols, K-dim 128 = 4x32 ----
    {
        v8f acc[4];
#pragma unroll
        for (int nt = 0; nt < 4; ++nt) acc[nt] = (v8f)(0.0f);
#pragma unroll
        for (int ks = 0; ks < 4; ++ks) {
            v16h a = ldsA(sKP + wv * 16 * (2 * BSZ_), 2 * BSZ_, lane, ks * 32);
#pragma unroll
            for (int nt = 0; nt < 4; ++nt) {
                v16h bb = ldsB(sVT + nt * 16 * (2 * BSZ_), 2 * BSZ_, lane, ks * 32);
                acc[nt] = __builtin_amdgcn_wmma_f32_16x16x32_f16(
                    false, a, false, bb, (short)0, acc[nt], false, false);
            }
        }
        __syncthreads();   // all softmax reads of sDO done; safe to overwrite as O
        int row0 = wv * 16 + (lane >> 4) * 8;
        int col0 = lane & 15;
#pragma unroll
        for (int nt = 0; nt < 4; ++nt)
#pragma unroll
            for (int r = 0; r < 8; ++r)
                sDO[(row0 + r) * ND_ + nt * 16 + col0] = acc[nt][r];
    }
    __syncthreads();

    // ---- scatter by token id (this IS the unsort) + normalize by sum ----
    if (tid < BSZ_) {
        int tq = sTq[tid];
        // recover 1/sum from lse and row max is folded in: p_unnorm = exp(x-mx),
        // o_norm = o_unnorm * exp(mx - lse) ... exp(mx - lse) = 1/sum
        float lse = sLse[tid];
        // mx = lse - log(sum); we stored unnormalized p, so divide by sum:
        // sum = exp(lse - mx). Instead recompute directly:
        // note: sLse = mx + log(sum)  ->  1/sum = exp(mx - sLse). We need mx:
        // simplest: re-derive sum from row of P? cheaper: store via lse trick:
        // o_final = o_unnorm * inv_sum, with inv_sum computed below.
        float inv_sum;
        {
            // recompute sum cheaply from stored f16 P row
            float s = 0.0f;
            for (int j = 0; j < 2 * BSZ_; ++j) s += (float)sKP[tid * (2 * BSZ_) + j];
            inv_sum = 1.0f / s;
        }
        size_t base = (((size_t)b * NH_ + h) * NS_ + tq) * ND_;
        float* dst = oh + base;
#pragma unroll
        for (int e = 0; e < ND_; ++e) dst[e] = sDO[tid * ND_ + e] * inv_sum;
        lseh[((size_t)b * NH_ + h) * NS_ + tq] = lse;
    }
}

// ------------------------------------------------------------------
// Kernel 4: combine hash rounds.  out = sum_h o_h * softmax_h(lse_h)
// grid (S/4, B), block 256 (4 tokens x 64 dims).
__global__ __launch_bounds__(256) void lsh_combine_kernel(
    const float* __restrict__ oh, const float* __restrict__ lseh,
    float* __restrict__ out) {
    const int tid = threadIdx.x;
    const int d = tid & 63;
    const int sub = tid >> 6;
    const int b = blockIdx.y;
    const int t = blockIdx.x * 4 + sub;
    float lg[NH_];
    float mx = -3.0e38f;
#pragma unroll
    for (int h = 0; h < NH_; ++h) {
        lg[h] = lseh[((size_t)b * NH_ + h) * NS_ + t];
        mx = fmaxf(mx, lg[h]);
    }
    float sum = 0.0f;
#pragma unroll
    for (int h = 0; h < NH_; ++h) { lg[h] = expf(lg[h] - mx); sum += lg[h]; }
    float inv = 1.0f / sum;
    float acc = 0.0f;
#pragma unroll
    for (int h = 0; h < NH_; ++h)
        acc = fmaf(oh[(((size_t)b * NH_ + h) * NS_ + t) * ND_ + d], lg[h] * inv, acc);
    out[((size_t)b * NS_ + t) * ND_ + d] = acc;
}

// ------------------------------------------------------------------
extern "C" void kernel_launch(void* const* d_in, const int* in_sizes, int n_in,
                              void* d_out, int out_size, void* d_ws, size_t ws_size,
                              hipStream_t stream) {
    const float* qk  = (const float*)d_in[0];
    const float* v   = (const float*)d_in[1];
    const float* rot = (const float*)d_in[2];
    float* out = (float*)d_out;

    char* ws = (char*)d_ws;
    const size_t nBHS = (size_t)NB_ * NH_ * NS_;
    int*   bkt  = (int*)(ws);
    int*   st   = (int*)(ws + nBHS * 4);
    float* lseh = (float*)(ws + 2 * nBHS * 4);
    float* oh   = (float*)(ws + 3 * nBHS * 4);   // B*H*S*D floats (128MB)

    lsh_hash_kernel<<<dim3(NS_ / 256, NB_), 256, 0, stream>>>(qk, rot, bkt);
    lsh_sort_kernel<<<dim3(NH_, NB_), 64, 0, stream>>>(bkt, st);
    lsh_attn_kernel<<<dim3(NCH_, NB_), 128, 0, stream>>>(qk, v, st, oh, lseh);
    lsh_combine_kernel<<<dim3(NS_ / 4, NB_), 256, 0, stream>>>(oh, lseh, out);
}
